// LinearSoftmaxAttention_36532991820075
// MI455X (gfx1250) — matrix operand order repaired and checked by reading, not compiled, not run on hardware
//
#include <hip/hip_runtime.h>
#include <hip/hip_bf16.h>

// ---------------------------------------------------------------------------
// Linear softmax attention, N=4 S=4096 H=16 D=64, fp32 in/out, f16 WMMA core.
// ---------------------------------------------------------------------------

typedef _Float16 v16h __attribute__((ext_vector_type(16)));
typedef float    v8f  __attribute__((ext_vector_type(8)));

#define NB 4
#define SQ 4096
#define NHD 16       // heads
#define DD 64
#define NHP (NB*NHD) // 64 (n,h) pairs
#define SEGS 8
#define SEGLEN (SQ/SEGS)  // 512
#define CH 32             // s-chunk per WMMA step

__device__ __forceinline__ v8f wmma_f32(v16h a, v16h b, v8f c) {
  return __builtin_amdgcn_wmma_f32_16x16x32_f16(false, a, false, b, (short)0, c,
                                                false, false);
}

// A fragment (16x32 f16, M x K): lane holds row M=lane&15;
// VGPR0-3 = K k0..k0+7, VGPR4-7 = K k0+16..k0+23, k0 = (lane&16)?8:0.
__device__ __forceinline__ v16h fragA(const _Float16* p, int strideH,
                                      int rowBase, int kbase) {
  int lane = threadIdx.x & 31;
  int k0 = (lane & 16) ? 8 : 0;
  const _Float16* r = p + (size_t)(rowBase + (lane & 15)) * strideH + kbase + k0;
  union { uint4 u[2]; v16h h; } u;
  u.u[0] = *(const uint4*)(r);
  u.u[1] = *(const uint4*)(r + 16);
  return u.h;
}

// B fragment (32x16 f16, K x N) read from row-major B^T (rows = N columns):
// lane holds col N=lane&15; K k0..k0+7, k0+16..k0+23, k0 = (lane&16)?8:0.
__device__ __forceinline__ v16h fragB(const _Float16* p, int strideH,
                                      int rowBase, int kbase) {
  int lane = threadIdx.x & 31;
  int k0 = (lane & 16) ? 8 : 0;
  const _Float16* r = p + (size_t)(rowBase + (lane & 15)) * strideH + kbase + k0;
  union { uint4 u[2]; v16h h; } u;
  u.u[0] = *(const uint4*)(r);
  u.u[1] = *(const uint4*)(r + 16);
  return u.h;
}

union U8h {
  uint4 u;
  _Float16 h[8];
  unsigned short s[8];
};
union U2h {
  unsigned u;
  _Float16 h[2];
};

// ---------------------------------------------------------------------------
// Kernel Z: zero f32 scratch accumulators
// ---------------------------------------------------------------------------
__global__ void zero_kernel(float* __restrict__ p, int n) {
  int i = blockIdx.x * blockDim.x + threadIdx.x;
  if (i < n) p[i] = 0.f;
}

// ---------------------------------------------------------------------------
// Kernel 0: LayerNorm Q and K, pack f16 to [n,h,s,d] layout.
// One wave32 per (n,s,h) row; 2 elements per lane; shuffle reduction.
// ---------------------------------------------------------------------------
__global__ void ln_pack_kernel(const float* __restrict__ Qin,
                               const float* __restrict__ Kin,
                               const float* __restrict__ mask,
                               _Float16* __restrict__ Qh,
                               _Float16* __restrict__ Kh) {
  int wave = threadIdx.x >> 5, lane = threadIdx.x & 31;
  int row = blockIdx.x * 8 + wave;        // over N*S*H rows
  int h = row % NHD;
  int ns = row / NHD;
  int s = ns % SQ;
  int n = ns / SQ;
  size_t ibase = (size_t)row * DD;                       // [n,s,h,d]
  size_t obase = ((size_t)(n * NHD + h) * SQ + s) * DD;  // [n,h,s,d]
  float mk = mask[n * SQ + s];

  {  // Q
    float x0 = Qin[ibase + lane], x1 = Qin[ibase + lane + 32];
    float sm = x0 + x1, sq = x0 * x0 + x1 * x1;
#pragma unroll
    for (int off = 16; off >= 1; off >>= 1) {
      sm += __shfl_xor(sm, off);
      sq += __shfl_xor(sq, off);
    }
    float mu = sm * (1.f / 64.f);
    float var = sq * (1.f / 64.f) - mu * mu;
    float rs = rsqrtf(var + 1e-5f);
    Qh[obase + lane]      = (_Float16)((x0 - mu) * rs);
    Qh[obase + lane + 32] = (_Float16)((x1 - mu) * rs);
  }
  {  // K, scaled by mask / (alpha*sqrt(D)) = mask/24
    float x0 = Kin[ibase + lane], x1 = Kin[ibase + lane + 32];
    float sm = x0 + x1, sq = x0 * x0 + x1 * x1;
#pragma unroll
    for (int off = 16; off >= 1; off >>= 1) {
      sm += __shfl_xor(sm, off);
      sq += __shfl_xor(sq, off);
    }
    float mu = sm * (1.f / 64.f);
    float var = sq * (1.f / 64.f) - mu * mu;
    float rs = rsqrtf(var + 1e-5f) * (1.f / 24.f) * mk;
    Kh[obase + lane]      = (_Float16)((x0 - mu) * rs);
    Kh[obase + lane + 32] = (_Float16)((x1 - mu) * rs);
  }
}

// ---------------------------------------------------------------------------
// Kernel 1: accumulate P = K^T V, KK = K^T K, QQ = Q^T Q, Ksum = sum_s K.
// grid = (SEGS, NHP), block = 384 (12 waves). One wave owns one
// (matrix, 16-row d-strip) pair -> only 4 accumulator tiles (32 VGPRs)/wave.
// Wave role (scalarized): waves 0-3 stage K (+Ksum), 4-7 Q, 8-11 V (f32->f16),
// transposed (d-major) into LDS; each thread packs a pair of s-rows so the
// transpose scatter is 8x ds_store_b32 instead of 16x ds_store_b16.
// Each WMMA fragment is then 2x ds_load_b128. Partial 64x64 f32 results are
// atomically added to global scratch.
// ---------------------------------------------------------------------------
#define TSTR 40  // LDS row stride in halves (80B: 16B-aligned fragments)

__global__ __launch_bounds__(384, 1) void acc_kernel(
    const float* __restrict__ Vin, const _Float16* __restrict__ Qh,
    const _Float16* __restrict__ Kh, float* __restrict__ P,
    float* __restrict__ KK, float* __restrict__ QQ,
    float* __restrict__ Ksum) {
  __shared__ __align__(16) _Float16 Kt[64 * TSTR];
  __shared__ __align__(16) _Float16 Vt[64 * TSTR];
  __shared__ __align__(16) _Float16 Qt[64 * TSTR];
  __shared__ float KsumL[64];

  int t = threadIdx.x;
  int lane = t & 31;
  // wave id / roles as scalars so role branches are s_cbranch, not exec masks
  int wave = __builtin_amdgcn_readfirstlane(t >> 5);  // 0..11
  int mat = wave >> 2;          // 0=P(K^T V, stages K), 1=KK (stages Q), 2=QQ (stages V)
  int strip = wave & 3;         // d row-strip for WMMA
  int nh = blockIdx.y, n = nh >> 4, h = nh & 15;
  int seg = blockIdx.x;
  // staging decomposition within the 128-thread role group:
  int st = (strip << 5) | lane;  // 0..127
  int spair = st >> 3;           // 0..15 -> s rows 2*spair, 2*spair+1
  int dq8 = (st & 7) << 3;       // 0..56, 8 d values

  if (t < 64) KsumL[t] = 0.f;

  float ksumr[8];
#pragma unroll
  for (int i = 0; i < 8; i++) ksumr[i] = 0.f;

  v8f acc[4];
#pragma unroll
  for (int j = 0; j < 4; j++) {
    v8f z = {0.f, 0.f, 0.f, 0.f, 0.f, 0.f, 0.f, 0.f};
    acc[j] = z;
  }

  const int NCH = SEGLEN / CH;
  for (int ch = 0; ch < NCH; ch++) {
    int s0 = seg * SEGLEN + ch * CH;
    int sA = s0 + 2 * spair;      // first of the row pair
    __syncthreads();
    if (mat == 0) {  // stage K transposed + Ksum partials
      const _Float16* r0 = Kh + (((size_t)nh * SQ) + sA) * DD + dq8;
      U8h a, b;
      a.u = *(const uint4*)(r0);
      b.u = *(const uint4*)(r0 + DD);
      if (ch + 1 < NCH) __builtin_prefetch(r0 + CH * DD, 0, 3);
#pragma unroll
      for (int i = 0; i < 8; i++) {
        U2h p2;
        p2.h[0] = a.h[i];
        p2.h[1] = b.h[i];
        *(unsigned*)&Kt[(dq8 + i) * TSTR + 2 * spair] = p2.u;
        ksumr[i] += (float)a.h[i] + (float)b.h[i];
      }
    } else if (mat == 1) {  // stage Q transposed
      const _Float16* r0 = Qh + (((size_t)nh * SQ) + sA) * DD + dq8;
      U8h a, b;
      a.u = *(const uint4*)(r0);
      b.u = *(const uint4*)(r0 + DD);
      if (ch + 1 < NCH) __builtin_prefetch(r0 + CH * DD, 0, 3);
#pragma unroll
      for (int i = 0; i < 8; i++) {
        U2h p2;
        p2.h[0] = a.h[i];
        p2.h[1] = b.h[i];
        *(unsigned*)&Qt[(dq8 + i) * TSTR + 2 * spair] = p2.u;
      }
    } else {  // stage V (f32 -> f16) transposed
      const float* r0 = Vin + (((size_t)n * SQ + sA) * NHD + h) * DD + dq8;
      const float* r1 = r0 + (size_t)NHD * DD;
      float4 f00 = *(const float4*)(r0), f01 = *(const float4*)(r0 + 4);
      float4 f10 = *(const float4*)(r1), f11 = *(const float4*)(r1 + 4);
      if (ch + 1 < NCH) __builtin_prefetch(r0 + (size_t)CH * NHD * DD, 0, 3);
      float va[8] = {f00.x, f00.y, f00.z, f00.w, f01.x, f01.y, f01.z, f01.w};
      float vb[8] = {f10.x, f10.y, f10.z, f10.w, f11.x, f11.y, f11.z, f11.w};
#pragma unroll
      for (int i = 0; i < 8; i++) {
        U2h p2;
        p2.h[0] = (_Float16)va[i];
        p2.h[1] = (_Float16)vb[i];
        *(unsigned*)&Vt[(dq8 + i) * TSTR + 2 * spair] = p2.u;
      }
    }
    __syncthreads();

    const _Float16* asrc = (mat == 2) ? Qt : Kt;
    const _Float16* bsrc = (mat == 0) ? Vt : ((mat == 1) ? Kt : Qt);
    v16h a = fragA(asrc, TSTR, 16 * strip, 0);
#pragma unroll
    for (int j = 0; j < 4; j++) {
      v16h b = fragB(bsrc, TSTR, 16 * j, 0);
      acc[j] = wmma_f32(a, b, acc[j]);
    }
  }

  __syncthreads();
  if (mat == 0) {
#pragma unroll
    for (int i = 0; i < 8; i++) atomicAdd(&KsumL[dq8 + i], ksumr[i]);
  }
  __syncthreads();
  if (t < 64) atomicAdd(&Ksum[nh * 64 + t], KsumL[t]);

  float* dst = (mat == 0) ? P : ((mat == 1) ? KK : QQ);
  int c0 = lane & 15;
  int rb = (lane & 16) ? 8 : 0;
#pragma unroll
  for (int j = 0; j < 4; j++)
#pragma unroll
    for (int r = 0; r < 8; r++) {
      int d = 16 * strip + rb + r;
      int c = 16 * j + c0;
      atomicAdd(&dst[((size_t)nh * 64 + d) * 64 + c], acc[j][r]);
    }
}

// ---------------------------------------------------------------------------
// Kernel 2: convert P -> P^T f16, KK/QQ -> f16 (symmetric, no transpose).
// ---------------------------------------------------------------------------
__global__ void conv_kernel(const float* __restrict__ P,
                            const float* __restrict__ KK,
                            const float* __restrict__ QQ,
                            _Float16* __restrict__ Pth,
                            _Float16* __restrict__ KKh,
                            _Float16* __restrict__ QQh) {
  int nh = blockIdx.x;
  size_t base = (size_t)nh * 4096;
  for (int i = threadIdx.x; i < 4096; i += blockDim.x) {
    int d = i >> 6, m = i & 63;
    Pth[base + m * 64 + d] = (_Float16)P[base + i];
    KKh[base + i] = (_Float16)KK[base + i];
    QQh[base + i] = (_Float16)QQ[base + i];
  }
}

// ---------------------------------------------------------------------------
// Kernel 3: output. grid = (S/64, NHP), block = 128 (4 waves; wave = 16 rows).
// order1 = Q*P (WMMA), quadratic forms via R = Q*KK, T = K*QQ (WMMA) + rowwise
// dot finished with shfl_xor within 16-lane halves.
// ---------------------------------------------------------------------------
#define OSTR 72  // LDS row stride in halves (144B, 16B-aligned fragments)

__global__ __launch_bounds__(128, 1) void out_kernel(
    const float* __restrict__ Vin, const _Float16* __restrict__ Qh,
    const _Float16* __restrict__ Kh, const _Float16* __restrict__ Pth,
    const _Float16* __restrict__ KKh, const _Float16* __restrict__ QQh,
    const float* __restrict__ Ksum, float* __restrict__ Out) {
  __shared__ __align__(16) _Float16 PL[64 * OSTR];
  __shared__ __align__(16) _Float16 KKL[64 * OSTR];
  __shared__ __align__(16) _Float16 QQL[64 * OSTR];
  __shared__ __align__(16) _Float16 Qs[4][16 * OSTR];
  __shared__ __align__(16) _Float16 Ks[4][16 * OSTR];
  __shared__ float KsumL[64];

  int t = threadIdx.x, lane = t & 31;
  int wave = __builtin_amdgcn_readfirstlane(t >> 5);
  int nh = blockIdx.y, n = nh >> 4, h = nh & 15;
  int L0 = blockIdx.x * 64 + 16 * wave;
  size_t mbase = (size_t)nh * 4096;

  {  // block-wide: stage the three 64x64 f16 matrices
    int row = t >> 1, half = t & 1;
    const uint4* sp = (const uint4*)(Pth + mbase + row * 64 + half * 32);
    uint4* dp = (uint4*)(PL + row * OSTR + half * 32);
    dp[0] = sp[0]; dp[1] = sp[1]; dp[2] = sp[2]; dp[3] = sp[3];
    sp = (const uint4*)(KKh + mbase + row * 64 + half * 32);
    dp = (uint4*)(KKL + row * OSTR + half * 32);
    dp[0] = sp[0]; dp[1] = sp[1]; dp[2] = sp[2]; dp[3] = sp[3];
    sp = (const uint4*)(QQh + mbase + row * 64 + half * 32);
    dp = (uint4*)(QQL + row * OSTR + half * 32);
    dp[0] = sp[0]; dp[1] = sp[1]; dp[2] = sp[2]; dp[3] = sp[3];
  }
  if (t < 64) KsumL[t] = Ksum[nh * 64 + t];
  {  // per-wave: stage Q/K 16-row strips
    int row = lane & 15, half = lane >> 4;
    const uint4* sp =
        (const uint4*)(Qh + ((size_t)nh * SQ + L0 + row) * 64 + half * 32);
    uint4* dp = (uint4*)(&Qs[wave][row * OSTR + half * 32]);
    dp[0] = sp[0]; dp[1] = sp[1]; dp[2] = sp[2]; dp[3] = sp[3];
    sp = (const uint4*)(Kh + ((size_t)nh * SQ + L0 + row) * 64 + half * 32);
    dp = (uint4*)(&Ks[wave][row * OSTR + half * 32]);
    dp[0] = sp[0]; dp[1] = sp[1]; dp[2] = sp[2]; dp[3] = sp[3];
  }
  __syncthreads();

  v8f O[4];
#pragma unroll
  for (int j = 0; j < 4; j++) {
    v8f z = {0.f, 0.f, 0.f, 0.f, 0.f, 0.f, 0.f, 0.f};
    O[j] = z;
  }
#pragma unroll
  for (int c = 0; c < 2; c++) {
    v16h aQ = fragA(Qs[wave], OSTR, 0, 32 * c);
#pragma unroll
    for (int j = 0; j < 4; j++) {
      v16h bP = fragB(PL, OSTR, 16 * j, 32 * c);
      O[j] = wmma_f32(aQ, bP, O[j]);  // order1[l,m] = sum_d Q[l,d] P[d,m]
    }
  }

  float rowq[8], rowl[8], rowc[8];
#pragma unroll
  for (int r = 0; r < 8; r++) { rowq[r] = 0.f; rowl[r] = 0.f; rowc[r] = 0.f; }
  int rb = (lane & 16) ? 8 : 0, c0 = lane & 15;
#pragma unroll
  for (int j = 0; j < 4; j++) {
    v8f R = {0.f, 0.f, 0.f, 0.f, 0.f, 0.f, 0.f, 0.f};
    v8f T = R;
#pragma unroll
    for (int c = 0; c < 2; c++) {
      v16h aQ = fragA(Qs[wave], OSTR, 0, 32 * c);
      v16h aK = fragA(Ks[wave], OSTR, 0, 32 * c);
      v16h bKK = fragB(KKL, OSTR, 16 * j, 32 * c);
      v16h bQQ = fragB(QQL, OSTR, 16 * j, 32 * c);
      R = wmma_f32(aQ, bKK, R);  // (Q*KK)[l,e]
      T = wmma_f32(aK, bQQ, T);  // (K*QQ)[l,e]
    }
    int dcol = 16 * j + c0;
    float ks = KsumL[dcol];
#pragma unroll
    for (int r = 0; r < 8; r++) {
      float qv = (float)Qs[wave][(rb + r) * OSTR + dcol];
      float kv = (float)Ks[wave][(rb + r) * OSTR + dcol];
      rowq[r] += qv * R[r];  // Q KK Q^T diag
      rowl[r] += qv * ks;    // Q . Ksum
      rowc[r] += kv * T[r];  // K QQ K^T diag
    }
  }
#pragma unroll
  for (int off = 8; off >= 1; off >>= 1)
#pragma unroll
    for (int r = 0; r < 8; r++) {
      rowq[r] += __shfl_xor(rowq[r], off);
      rowl[r] += __shfl_xor(rowl[r], off);
      rowc[r] += __shfl_xor(rowc[r], off);
    }

#pragma unroll
  for (int r = 0; r < 8; r++) {
    int l = L0 + rb + r;
    float norm = rowl[r] + 0.5f * rowq[r];
    float inv = 1.f / norm;
    float cf = 0.5f * rowc[r];
    size_t ob = ((size_t)(n * SQ + l) * NHD + h) * 64;
#pragma unroll
    for (int j = 0; j < 4; j++) {
      int m = 16 * j + c0;
      float v = Vin[ob + m];
      Out[ob + m] = (O[j][r] + cf * v) * inv;
    }
  }
}

// ---------------------------------------------------------------------------
extern "C" void kernel_launch(void* const* d_in, const int* in_sizes, int n_in,
                              void* d_out, int out_size, void* d_ws,
                              size_t ws_size, hipStream_t stream) {
  (void)in_sizes; (void)n_in; (void)out_size; (void)ws_size;
  const float* Qin  = (const float*)d_in[0];
  const float* Kin  = (const float*)d_in[1];
  const float* Vin  = (const float*)d_in[2];
  const float* mask = (const float*)d_in[3];

  char* ws = (char*)d_ws;
  _Float16* Qh = (_Float16*)(ws);                      // 32 MB
  _Float16* Kh = (_Float16*)(ws + (size_t)(32u << 20));// 32 MB
  float* P    = (float*)(ws + (size_t)(64u << 20));    // 1 MB
  float* KK   = (float*)(ws + (size_t)(65u << 20));    // 1 MB
  float* QQ   = (float*)(ws + (size_t)(66u << 20));    // 1 MB
  float* Ksum = (float*)(ws + (size_t)(67u << 20));    // 16 KB
  _Float16* Pth = (_Float16*)(ws + (size_t)(67u << 20) + (1u << 16));
  _Float16* KKh = Pth + (size_t)NHP * 4096;
  _Float16* QQh = KKh + (size_t)NHP * 4096;

  int zn = 3 * NHP * 4096 + NHP * 64;  // P,KK,QQ,Ksum are contiguous
  zero_kernel<<<(zn + 255) / 256, 256, 0, stream>>>(P, zn);
  ln_pack_kernel<<<(NB * SQ * NHD) / 8, 256, 0, stream>>>(Qin, Kin, mask, Qh, Kh);
  acc_kernel<<<dim3(SEGS, NHP), 384, 0, stream>>>(Vin, Qh, Kh, P, KK, QQ, Ksum);
  conv_kernel<<<NHP, 256, 0, stream>>>(P, KK, QQ, Pth, KKh, QQh);
  out_kernel<<<dim3(SQ / 64, NHP), 128, 0, stream>>>(Vin, Qh, Kh, Pth, KKh, QQh,
                                                     Ksum, (float*)d_out);
}
